// GCNEncoder_74990128988468
// MI455X (gfx1250) — matrix operand order repaired and checked by reading, not compiled, not run on hardware
//
#include <hip/hip_runtime.h>
#include <math.h>

typedef __attribute__((ext_vector_type(16))) __bf16 v16bf;
typedef __attribute__((ext_vector_type(8)))  float  v8f;

union V16U { v16bf v; uint4 q[2]; };

__device__ __forceinline__ float gelu_f(float x) {
    // tanh-approx GELU (jax.nn.gelu approximate=True)
    float u = 0.7978845608028654f * (x + 0.044715f * x * x * x);
    float e = __expf(2.0f * u);
    float t = (e - 1.0f) / (e + 1.0f);
    return 0.5f * x * (1.0f + t);
}

__device__ __forceinline__ v8f wmma_bf16(v16bf a, v16bf b, v8f c) {
    return __builtin_amdgcn_wmma_f32_16x16x32_bf16(false, a, false, b, (short)0, c, false, false);
}

// A fragment: rows m..m+15, K columns kbase..kbase+31, from row-major [*, stride] bf16.
// lane<16: K offsets {0..7,16..23}; lane>=16: {8..15,24..31}  (ISA 16-bit A layout)
__device__ __forceinline__ v16bf load_frag_A(const __bf16* base, int stride, int m, int kbase, int lane) {
    int mm  = m + (lane & 15);
    int off = (lane < 16) ? 0 : 8;
    const __bf16* p = base + (size_t)mm * stride + kbase + off;
    V16U t;
    t.q[0] = *(const uint4*)(p);
    t.q[1] = *(const uint4*)(p + 16);
    return t.v;
}

// B fragment from transposed weights Wt[N][K]: lane -> column n = nbase + lane%16,
// lane<16: K offsets {0..7,16..23}; lane>=16: {8..15,24..31}  (matches A layout)
__device__ __forceinline__ v16bf load_frag_B(const __bf16* Wt, int K, int nbase, int kbase, int lane) {
    int n = nbase + (lane & 15);
    int k = kbase + ((lane >> 4) << 3);
    const __bf16* p = Wt + (size_t)n * K + k;
    V16U t;
    t.q[0] = *(const uint4*)(p);
    t.q[1] = *(const uint4*)(p + 16);
    return t.v;
}

// ---------------------------------------------------------------------------
// Fused 4-layer MLP: KIN -> 1024 -> 2048 -> 1024 -> 128, GELU between layers.
// One workgroup (8 waves) processes 32 rows. Hidden-2048 is chunked by 256 so
// it never materializes; layer-3 accumulators persist in registers.
// IS_EDGE: builds x = [attr[src] | RBF(dist)] and atomic-adds output to agg[dst].
// else:    x = attr[row], output stored to out[row].
// ---------------------------------------------------------------------------
template <int KIN, bool IS_EDGE>
__global__ __launch_bounds__(256, 1)
void k_fused_mlp(const __bf16* __restrict__ attr,      // [N,256] bf16
                 const float* __restrict__ coord,      // [N,3]
                 const int* __restrict__ esrc,
                 const int* __restrict__ edst,
                 int count,
                 const __bf16* __restrict__ W0t, const float* __restrict__ b0,  // [1024][KIN]
                 const __bf16* __restrict__ W1t, const float* __restrict__ b1,  // [2048][1024]
                 const __bf16* __restrict__ W2t, const float* __restrict__ b2,  // [1024][2048]
                 const __bf16* __restrict__ W3t, const float* __restrict__ b3,  // [128][1024]
                 float* __restrict__ outp)              // agg[N,128] or dst_attr[N,128]
{
    constexpr int M   = 32;
    constexpr int H1  = 1024;
    constexpr int CW  = 256;                       // h2 chunk width
    constexpr int SX  = M * KIN * 2;
    constexpr int SMEM_BYTES = SX + M * H1 * 2 + M * CW * 2 + 512;
    __shared__ alignas(16) unsigned char smem[SMEM_BYTES];

    __bf16* sx   = (__bf16*)(smem);
    __bf16* sh1  = (__bf16*)(smem + SX);                 // h1, later reused for h3
    __bf16* sh2  = (__bf16*)(smem + SX + M * H1 * 2);    // 256-col chunk of h2
    int*    ssrc = (int*)  (smem + SX + M * H1 * 2 + M * CW * 2);
    int*    sdid = ssrc + M;
    float*  sdst = (float*)(sdid + M);                   // distances

    const int tid  = threadIdx.x;
    const int lane = tid & 31;
    const int wave = tid >> 5;
    const int base = blockIdx.x * M;

    // ---- Phase 0: build x tile --------------------------------------------
    if (IS_EDGE) {
        if (tid < M) {
            int e = base + tid;
            int s = (e < count) ? esrc[e] : 0;
            int d = (e < count) ? edst[e] : 0;
            ssrc[tid] = s;
            sdid[tid] = d;
            float dx = coord[3 * s + 0] - coord[3 * d + 0];
            float dy = coord[3 * s + 1] - coord[3 * d + 1];
            float dz = coord[3 * s + 2] - coord[3 * d + 2];
            sdst[tid] = sqrtf(dx * dx + dy * dy + dz * dz);
        }
        __syncthreads();
        // attr part: cols [0,256): copy as uint2 (4 bf16 each)
        for (int idx = tid; idx < M * 64; idx += 256) {
            int i = idx >> 6, j4 = idx & 63;
            uint2 v = make_uint2(0u, 0u);
            if (base + i < count)
                v = ((const uint2*)(attr + (size_t)ssrc[i] * 256))[j4];
            ((uint2*)(sx + (size_t)i * KIN))[j4] = v;
        }
        // RBF part: cols [256,512)
        const float inv_sigma = 51.0f;         // 1/(5/255)
        const float dmu = 5.0f / 255.0f;
        for (int idx = tid; idx < M * 256; idx += 256) {
            int i = idx >> 8, j = idx & 255;
            float v = 0.0f;
            if (base + i < count) {
                float t = (sdst[i] - (float)j * dmu) * inv_sigma;
                v = __expf(-0.5f * t * t);
            }
            sx[(size_t)i * KIN + 256 + j] = (__bf16)v;
        }
    } else {
        for (int idx = tid; idx < M * 64; idx += 256) {
            int i = idx >> 6, j4 = idx & 63;
            uint2 v = make_uint2(0u, 0u);
            if (base + i < count)
                v = ((const uint2*)(attr + (size_t)(base + i) * 256))[j4];
            ((uint2*)(sx + (size_t)i * KIN))[j4] = v;
        }
    }
    __syncthreads();

    // ---- Layer 1: h1 = gelu(x @ W0 + b0)  [32 x 1024] ---------------------
    {
        v8f acc[2][8];
        for (int a = 0; a < 2; ++a)
            for (int b = 0; b < 8; ++b)
                for (int r = 0; r < 8; ++r) acc[a][b][r] = 0.0f;
        constexpr int KS = KIN / 32;
        for (int ks = 0; ks < KS; ++ks) {
            int kb = ks * 32;
            v16bf a0 = load_frag_A(sx, KIN, 0,  kb, lane);
            v16bf a1 = load_frag_A(sx, KIN, 16, kb, lane);
            #pragma unroll
            for (int nt = 0; nt < 8; ++nt) {
                v16bf b = load_frag_B(W0t, KIN, wave * 128 + nt * 16, kb, lane);
                acc[0][nt] = wmma_bf16(a0, b, acc[0][nt]);
                acc[1][nt] = wmma_bf16(a1, b, acc[1][nt]);
            }
        }
        for (int mt = 0; mt < 2; ++mt)
            for (int nt = 0; nt < 8; ++nt) {
                int col = wave * 128 + nt * 16 + (lane & 15);
                float bb = b0[col];
                #pragma unroll
                for (int r = 0; r < 8; ++r) {
                    int m = mt * 16 + ((lane >> 4) << 3) + r;
                    sh1[(size_t)m * H1 + col] = (__bf16)gelu_f(acc[mt][nt][r] + bb);
                }
            }
    }
    __syncthreads();

    // ---- Layers 2+3 fused over 256-col chunks of hidden-2048 --------------
    // Each wave owns 2 N-tiles of the chunk in layer2 (4 WMMAs per A-pair)
    // and 8 N-tiles of the 1024-wide layer-3 output (persistent acc3).
    v8f acc3[2][8];
    for (int a = 0; a < 2; ++a)
        for (int b = 0; b < 8; ++b)
            for (int r = 0; r < 8; ++r) acc3[a][b][r] = 0.0f;

    for (int c = 0; c < 8; ++c) {
        // layer2: this wave computes cols [c*256 + wave*32, +32) of h2
        v8f a2[2][2];
        for (int a = 0; a < 2; ++a)
            for (int b = 0; b < 2; ++b)
                for (int r = 0; r < 8; ++r) a2[a][b][r] = 0.0f;
        for (int ks = 0; ks < 32; ++ks) {
            int kb = ks * 32;
            v16bf f0 = load_frag_A(sh1, H1, 0,  kb, lane);
            v16bf f1 = load_frag_A(sh1, H1, 16, kb, lane);
            #pragma unroll
            for (int nt = 0; nt < 2; ++nt) {
                v16bf b = load_frag_B(W1t, 1024, c * CW + wave * 32 + nt * 16, kb, lane);
                a2[0][nt] = wmma_bf16(f0, b, a2[0][nt]);
                a2[1][nt] = wmma_bf16(f1, b, a2[1][nt]);
            }
        }
        __syncthreads();  // previous chunk's layer-3 reads of sh2 are done
        for (int nt = 0; nt < 2; ++nt) {
            int coll = wave * 32 + nt * 16 + (lane & 15);
            float bb = b1[c * CW + coll];
            for (int mt = 0; mt < 2; ++mt) {
                #pragma unroll
                for (int r = 0; r < 8; ++r) {
                    int m = mt * 16 + ((lane >> 4) << 3) + r;
                    sh2[(size_t)m * CW + coll] = (__bf16)gelu_f(a2[mt][nt][r] + bb);
                }
            }
        }
        __syncthreads();
        // layer3 partial: acc3 += h2_chunk @ W2[chunk_rows, :]
        for (int ks = 0; ks < 8; ++ks) {
            int kb = ks * 32;
            v16bf f0 = load_frag_A(sh2, CW, 0,  kb, lane);
            v16bf f1 = load_frag_A(sh2, CW, 16, kb, lane);
            int kg = c * CW + kb;
            #pragma unroll
            for (int nt = 0; nt < 8; ++nt) {
                v16bf b = load_frag_B(W2t, 2048, wave * 128 + nt * 16, kg, lane);
                acc3[0][nt] = wmma_bf16(f0, b, acc3[0][nt]);
                acc3[1][nt] = wmma_bf16(f1, b, acc3[1][nt]);
            }
        }
    }
    __syncthreads();
    // h3 = gelu(acc3 + b2) -> reuse sh1 region
    for (int mt = 0; mt < 2; ++mt)
        for (int nt = 0; nt < 8; ++nt) {
            int col = wave * 128 + nt * 16 + (lane & 15);
            float bb = b2[col];
            #pragma unroll
            for (int r = 0; r < 8; ++r) {
                int m = mt * 16 + ((lane >> 4) << 3) + r;
                sh1[(size_t)m * H1 + col] = (__bf16)gelu_f(acc3[mt][nt][r] + bb);
            }
        }
    __syncthreads();

    // ---- Layer 4: y = h3 @ W3 + b3  [32 x 128] ----------------------------
    {
        v8f a4[2];
        for (int a = 0; a < 2; ++a)
            for (int r = 0; r < 8; ++r) a4[a][r] = 0.0f;
        for (int ks = 0; ks < 32; ++ks) {
            int kb = ks * 32;
            v16bf f0 = load_frag_A(sh1, H1, 0,  kb, lane);
            v16bf f1 = load_frag_A(sh1, H1, 16, kb, lane);
            v16bf b  = load_frag_B(W3t, 1024, wave * 16, kb, lane);
            a4[0] = wmma_bf16(f0, b, a4[0]);
            a4[1] = wmma_bf16(f1, b, a4[1]);
        }
        int col = wave * 16 + (lane & 15);
        float bb = b3[col];
        for (int mt = 0; mt < 2; ++mt) {
            #pragma unroll
            for (int r = 0; r < 8; ++r) {
                int m   = mt * 16 + ((lane >> 4) << 3) + r;
                int row = base + m;
                if (row < count) {
                    float v = a4[mt][r] + bb;
                    if (IS_EDGE)
                        atomicAdd(&outp[(size_t)sdid[m] * 128 + col], v);
                    else
                        outp[(size_t)row * 128 + col] = v;
                }
            }
        }
    }
}

// ---------------------------------------------------------------------------
// Utility kernels
// ---------------------------------------------------------------------------

// Wt[n*K + k] = bf16( W[(k+rowoff)*N + n] )   (weights -> transposed bf16)
__global__ void k_transpose_bf16(const float* __restrict__ W, __bf16* __restrict__ Wt,
                                 int K, int N, int rowoff) {
    size_t idx = (size_t)blockIdx.x * 256 + threadIdx.x;
    size_t total = (size_t)K * N;
    if (idx >= total) return;
    int k = (int)(idx / N);
    int n = (int)(idx % N);
    Wt[(size_t)n * K + k] = (__bf16)W[(size_t)(k + rowoff) * N + n];
}

__global__ void k_embed_bf16(const float* __restrict__ table, const int* __restrict__ types,
                             __bf16* __restrict__ attr, int N) {
    int idx = blockIdx.x * 256 + threadIdx.x;
    if (idx >= N * 256) return;
    int n = idx >> 8, j = idx & 255;
    attr[idx] = (__bf16)table[(size_t)types[n] * 256 + j];
}

__global__ void k_zero_f32(float* __restrict__ p, int n) {
    int idx = blockIdx.x * 256 + threadIdx.x;
    if (idx < n) p[idx] = 0.0f;
}

// y[j] = (gelu?)( b[j] + sum_k x[k] * W[k*Nn + j] );  K==0 -> y = gelu?(b)
__global__ void k_vec_mm(const float* __restrict__ x, const float* __restrict__ W,
                         const float* __restrict__ b, float* __restrict__ y,
                         int K, int Nn, int apply_gelu) {
    int j = blockIdx.x * 256 + threadIdx.x;
    if (j >= Nn) return;
    float s = b[j];
    for (int k = 0; k < K; ++k) s += x[k] * W[(size_t)k * Nn + j];
    y[j] = apply_gelu ? gelu_f(s) : s;
}

// out[n,j] = bfc[j] + cat(dst_attr[n], dst_feat, agg[n]) . Wfc[:,j]
__global__ void k_final(const float* __restrict__ dst_attr, const float* __restrict__ dst_feat,
                        const float* __restrict__ agg, const float* __restrict__ Wfc,
                        const float* __restrict__ bfc, float* __restrict__ out, int N) {
    int idx = blockIdx.x * 256 + threadIdx.x;
    if (idx >= N * 128) return;
    int n = idx >> 7, j = idx & 127;
    float s = bfc[j];
    const float* a0 = dst_attr + (size_t)n * 128;
    const float* a2 = agg + (size_t)n * 128;
    for (int k = 0; k < 128; ++k) s += a0[k]       * Wfc[(size_t)k * 128 + j];
    for (int k = 0; k < 128; ++k) s += dst_feat[k] * Wfc[(size_t)(128 + k) * 128 + j];
    for (int k = 0; k < 128; ++k) s += a2[k]       * Wfc[(size_t)(256 + k) * 128 + j];
    out[idx] = s;
}

// ---------------------------------------------------------------------------
extern "C" void kernel_launch(void* const* d_in, const int* in_sizes, int n_in,
                              void* d_out, int out_size, void* d_ws, size_t ws_size,
                              hipStream_t stream) {
    // --- classify inputs by size pattern (robust to pytree flatten order) ---
    int iTypes = -1, iCoord = -1, iEdge = -1, iEmb = -1, iS = -1, iD = -1, iF = -1, iFC = -1;
    for (int i = 0; i < n_in; ++i) {
        int sz = in_sizes[i];
        if      (sz == 30000)  iCoord = i;
        else if (sz == 320000) iEdge = i;
        else if (sz == 25600)  iEmb = i;
        else if (sz == 655360) iS = i;
        else if (sz == 262144) iD = i;
        else if (sz == 49152)  iFC = i;
        else if (sz == 131072 && i + 2 < n_in && in_sizes[i + 1] == 1024 &&
                 in_sizes[i + 2] == 2097152) iF = i;
        else if (sz == 10000 && iTypes < 0) iTypes = i;
    }
    const int*   types = (const int*)d_in[iTypes];
    const float* coord = (const float*)d_in[iCoord];
    const int*   eidx  = (const int*)d_in[iEdge];
    const float* embed = (const float*)d_in[iEmb];
    const float *sW[4], *sB[4], *dW[4], *dB[4], *fW[4], *fB[4];
    for (int l = 0; l < 4; ++l) {
        sW[l] = (const float*)d_in[iS + 2 * l]; sB[l] = (const float*)d_in[iS + 2 * l + 1];
        dW[l] = (const float*)d_in[iD + 2 * l]; dB[l] = (const float*)d_in[iD + 2 * l + 1];
        fW[l] = (const float*)d_in[iF + 2 * l]; fB[l] = (const float*)d_in[iF + 2 * l + 1];
    }
    const float* fcW = (const float*)d_in[iFC];
    const float* fcB = (const float*)d_in[iFC + 1];

    const int N = in_sizes[iTypes];
    const int E = in_sizes[iEdge] / 2;
    const int* esrc = eidx;
    const int* edst = eidx + E;

    // --- workspace layout ---------------------------------------------------
    char* ws = (char*)d_ws;
    size_t off = 0;
    auto alloc = [&](size_t bytes) { char* p = ws + off; off = (off + bytes + 255) & ~(size_t)255; return p; };
    __bf16* sW0t = (__bf16*)alloc((size_t)1024 * 512 * 2);
    __bf16* sW1t = (__bf16*)alloc((size_t)2048 * 1024 * 2);
    __bf16* sW2t = (__bf16*)alloc((size_t)1024 * 2048 * 2);
    __bf16* sW3t = (__bf16*)alloc((size_t)128 * 1024 * 2);
    __bf16* dW0t = (__bf16*)alloc((size_t)1024 * 256 * 2);
    __bf16* dW1t = (__bf16*)alloc((size_t)2048 * 1024 * 2);
    __bf16* dW2t = (__bf16*)alloc((size_t)1024 * 2048 * 2);
    __bf16* dW3t = (__bf16*)alloc((size_t)128 * 1024 * 2);
    __bf16* attr = (__bf16*)alloc((size_t)N * 256 * 2);
    float* dst_attr = (float*)alloc((size_t)N * 128 * 4);
    float* agg      = (float*)alloc((size_t)N * 128 * 4);
    float* h1f = (float*)alloc(1024 * 4);
    float* h2f = (float*)alloc(2048 * 4);
    float* h3f = (float*)alloc(1024 * 4);
    float* dfv = (float*)alloc(128 * 4);
    (void)ws_size;

    auto blk = [](size_t total) { return (unsigned)((total + 255) / 256); };

    // --- weight conversion (src W0 drops the 128 zero input rows) -----------
    k_transpose_bf16<<<blk((size_t)512 * 1024), 256, 0, stream>>>(sW[0], sW0t, 512, 1024, 128);
    k_transpose_bf16<<<blk((size_t)1024 * 2048), 256, 0, stream>>>(sW[1], sW1t, 1024, 2048, 0);
    k_transpose_bf16<<<blk((size_t)2048 * 1024), 256, 0, stream>>>(sW[2], sW2t, 2048, 1024, 0);
    k_transpose_bf16<<<blk((size_t)1024 * 128), 256, 0, stream>>>(sW[3], sW3t, 1024, 128, 0);
    k_transpose_bf16<<<blk((size_t)256 * 1024), 256, 0, stream>>>(dW[0], dW0t, 256, 1024, 0);
    k_transpose_bf16<<<blk((size_t)1024 * 2048), 256, 0, stream>>>(dW[1], dW1t, 1024, 2048, 0);
    k_transpose_bf16<<<blk((size_t)2048 * 1024), 256, 0, stream>>>(dW[2], dW2t, 2048, 1024, 0);
    k_transpose_bf16<<<blk((size_t)1024 * 128), 256, 0, stream>>>(dW[3], dW3t, 1024, 128, 0);

    // --- node embedding gather + agg zero -----------------------------------
    k_embed_bf16<<<blk((size_t)N * 256), 256, 0, stream>>>(embed, types, attr, N);
    k_zero_f32<<<blk((size_t)N * 128), 256, 0, stream>>>(agg, N * 128);

    // --- dst_feat = feat_fc(0): single vector via biases ---------------------
    k_vec_mm<<<blk(1024), 256, 0, stream>>>(fB[0], fW[1], fB[0], h1f, 0, 1024, 1);
    k_vec_mm<<<blk(2048), 256, 0, stream>>>(h1f, fW[1], fB[1], h2f, 1024, 2048, 1);
    k_vec_mm<<<blk(1024), 256, 0, stream>>>(h2f, fW[2], fB[2], h3f, 2048, 1024, 1);
    k_vec_mm<<<blk(128), 256, 0, stream>>>(h3f, fW[3], fB[3], dfv, 1024, 128, 0);

    // --- fused WMMA MLPs: per-node dst_fc, per-edge src_fc (+ scatter-add) ---
    k_fused_mlp<256, false><<<(N + 31) / 32, 256, 0, stream>>>(
        attr, coord, esrc, edst, N,
        dW0t, dB[0], dW1t, dB[1], dW2t, dB[2], dW3t, dB[3], dst_attr);
    k_fused_mlp<512, true><<<(E + 31) / 32, 256, 0, stream>>>(
        attr, coord, esrc, edst, E,
        sW0t, sB[0], sW1t, sB[1], sW2t, sB[2], sW3t, sB[3], agg);

    // --- final concat @ fc ---------------------------------------------------
    k_final<<<blk((size_t)N * 128), 256, 0, stream>>>(dst_attr, dfv, agg, fcW, fcB,
                                                      (float*)d_out, N);
}
